// UMGMQuantizer_82248623718638
// MI455X (gfx1250) — matrix operand order repaired and checked by reading, not compiled, not run on hardware
//
#include <hip/hip_runtime.h>
#include <hip/hip_bf16.h>

#define MM   2
#define KK   4096
#define DD   128
#define HWSZ 1024
#define NPOS 4096      // N*H*W = 4*32*32
#define CCH  256
#define NN   4
#define SL   16        // K-slices per (m, position-pair)
#define KTPS (256 / SL) // k-tiles (of 16) per slice

typedef __attribute__((ext_vector_type(16))) __bf16 v16bf;
typedef __attribute__((ext_vector_type(8)))  float  v8f;

union V16 { v16bf v; uint4 q[2]; };

__device__ __forceinline__ unsigned short f32_to_bf16(float f) {
    unsigned int u = __float_as_uint(f);
    unsigned int r = (u + 0x7FFFu + ((u >> 16) & 1u)) >> 16;
    return (unsigned short)r;
}
__device__ __forceinline__ float bf16_to_f32(unsigned short h) {
    return __uint_as_float(((unsigned int)h) << 16);
}

// ---------------- small precompute kernels (f32 VALU) ----------------

// Cm[o][c] = sum_j A[o][j] * B[j][c]    (256x256 = 256x256 * 256x256)
__global__ __launch_bounds__(256) void k_matmul256(const float* __restrict__ A,
                                                   const float* __restrict__ B,
                                                   float* __restrict__ Cm) {
    int o = blockIdx.x, c = threadIdx.x;
    float acc = 0.f;
    #pragma unroll 4
    for (int j = 0; j < CCH; ++j) acc = fmaf(A[o * CCH + j], B[j * CCH + c], acc);
    Cm[o * CCH + c] = acc;
}

// qb2[j] = qh_w[j,:] . enc_b + qh_b[j]
__global__ void k_qb2(const float* __restrict__ qh_w, const float* __restrict__ enc_b,
                      const float* __restrict__ qh_b, float* __restrict__ qb2) {
    int j = blockIdx.x * blockDim.x + threadIdx.x;
    if (j >= CCH) return;
    float acc = qh_b[j];
    for (int c = 0; c < CCH; ++c) acc = fmaf(qh_w[j * CCH + c], enc_b[c], acc);
    qb2[j] = acc;
}

// WfT[m][c][d] = sum_j pre_w[m][d][j] * T[(m*128+j)][c]
__global__ __launch_bounds__(128) void k_wft(const float* __restrict__ pre_w,
                                             const float* __restrict__ T_,
                                             float* __restrict__ WfT) {
    int bid = blockIdx.x;           // m*256 + c
    int m = bid >> 8, c = bid & 255, d = threadIdx.x;
    float acc = 0.f;
    for (int j = 0; j < DD; ++j)
        acc = fmaf(pre_w[((size_t)(m * DD + d)) * DD + j], T_[(m * DD + j) * CCH + c], acc);
    WfT[((size_t)(m * CCH + c)) * DD + d] = acc;
}

// bfv[m*128+d] = pre_w[m][d][:] . qb2[m-block] + pre_b[m][d]
__global__ void k_bfv(const float* __restrict__ pre_w, const float* __restrict__ qb2,
                      const float* __restrict__ pre_b, float* __restrict__ bfv) {
    int i = blockIdx.x * blockDim.x + threadIdx.x;
    if (i >= MM * DD) return;
    int m = i >> 7;
    float acc = pre_b[i];
    for (int j = 0; j < DD; ++j) acc = fmaf(pre_w[(size_t)i * DD + j], qb2[m * DD + j], acc);
    bfv[i] = acc;
}

// mapped_q[m][k][c] = codebook[m][k][:] . wC[m][c][:]  -> bf16 hi/lo split
__global__ __launch_bounds__(128) void k_mapq(const float* __restrict__ codebook,
                                              const float* __restrict__ wC,
                                              unsigned short* __restrict__ mq_hi,
                                              unsigned short* __restrict__ mq_lo) {
    __shared__ float cb[DD];
    int bid = blockIdx.x;           // m*4096 + k
    int m = bid >> 12, k = bid & 4095, t = threadIdx.x;   // t = c
    cb[t] = codebook[((size_t)(m * KK + k)) * DD + t];
    __syncthreads();
    const float* w = wC + ((size_t)(m * DD + t)) * DD;
    float acc = 0.f;
    #pragma unroll 4
    for (int d = 0; d < DD; ++d) acc = fmaf(w[d], cb[d], acc);
    unsigned short hi = f32_to_bf16(acc);
    size_t o = ((size_t)(m * KK + k)) * DD + t;
    mq_hi[o] = hi;
    mq_lo[o] = f32_to_bf16(acc - bf16_to_f32(hi));
}

// mapped_dq[m][k][c] (f32, for table precompute)
__global__ __launch_bounds__(128) void k_mapdq(const float* __restrict__ codebook,
                                               const float* __restrict__ wC,
                                               float* __restrict__ mdq) {
    __shared__ float cb[DD];
    int bid = blockIdx.x;
    int m = bid >> 12, k = bid & 4095, t = threadIdx.x;
    cb[t] = codebook[((size_t)(m * KK + k)) * DD + t];
    __syncthreads();
    const float* w = wC + ((size_t)(m * DD + t)) * DD;
    float acc = 0.f;
    #pragma unroll 4
    for (int d = 0; d < DD; ++d) acc = fmaf(w[d], cb[d], acc);
    mdq[((size_t)(m * KK + k)) * DD + t] = acc;
}

// c2[m][k] = ||codebook[m][k]||^2 ; expT = exp(logT)
__global__ void k_c2(const float* __restrict__ codebook, const float* __restrict__ logT,
                     float* __restrict__ c2, float* __restrict__ expT) {
    int i = blockIdx.x * blockDim.x + threadIdx.x;
    if (i >= MM * KK) return;
    const float* row = codebook + (size_t)i * DD;
    float s = 0.f;
    for (int d = 0; d < DD; ++d) s = fmaf(row[d], row[d], s);
    c2[i] = s;
    expT[i] = expf(logT[i]);
}

// G[m][o][d] = sum_e Wback[o][m*128+e] * post_w[m][e][d]
__global__ __launch_bounds__(128) void k_G(const float* __restrict__ Wback,
                                           const float* __restrict__ post_w,
                                           float* __restrict__ G) {
    int bid = blockIdx.x;           // m*256 + o
    int m = bid >> 8, o = bid & 255, d = threadIdx.x;
    float acc = 0.f;
    for (int e = 0; e < DD; ++e)
        acc = fmaf(Wback[o * CCH + m * DD + e], post_w[((size_t)(m * DD + e)) * DD + d], acc);
    G[((size_t)(m * CCH + o)) * DD + d] = acc;
}

// table[m][k][o] = G[m][o][:] . mdq[m][k][:]
__global__ __launch_bounds__(256) void k_table(const float* __restrict__ G,
                                               const float* __restrict__ mdq,
                                               float* __restrict__ table) {
    __shared__ float row[DD];
    int bid = blockIdx.x;           // m*4096 + k
    int m = bid >> 12, k = bid & 4095, o = threadIdx.x;
    if (o < DD) row[o] = mdq[((size_t)(m * KK + k)) * DD + o];
    __syncthreads();
    const float* g = G + ((size_t)(m * CCH + o)) * DD;
    float acc = 0.f;
    #pragma unroll 4
    for (int d = 0; d < DD; ++d) acc = fmaf(g[d], row[d], acc);
    table[((size_t)(m * KK + k)) * CCH + o] = acc;
}

// btot[o] = Wback[o][:] . post_b_flat + rest_w[o][:] . dqh_b + rest_b[o]
__global__ void k_btot(const float* __restrict__ Wback, const float* __restrict__ post_b,
                       const float* __restrict__ rest_w, const float* __restrict__ dqh_b,
                       const float* __restrict__ rest_b, float* __restrict__ btot) {
    int o = blockIdx.x * blockDim.x + threadIdx.x;
    if (o >= CCH) return;
    float acc = rest_b[o];
    for (int c = 0; c < CCH; ++c) acc = fmaf(Wback[o * CCH + c], post_b[c], acc);
    for (int j = 0; j < CCH; ++j) acc = fmaf(rest_w[o * CCH + j], dqh_b[j], acc);
    btot[o] = acc;
}

// Front fused GEMM: pre[m][pos][d] = WfT[m][:,d] . x_pix + bfv ; also x2 + bf16 split.
__global__ __launch_bounds__(128) void k_front(const float* __restrict__ x,
                                               const float* __restrict__ WfT,
                                               const float* __restrict__ bfv,
                                               unsigned short* __restrict__ pre_hi,
                                               unsigned short* __restrict__ pre_lo,
                                               float* __restrict__ x2) {
    __shared__ float xs[CCH];
    __shared__ float red[128];
    int bid = blockIdx.x;           // m*4096 + pos
    int m = bid >> 12, pos = bid & 4095;
    int n = pos >> 10, hw = pos & (HWSZ - 1);
    int t = threadIdx.x;            // t = d
    xs[t]       = x[((size_t)(n * CCH + t)) * HWSZ + hw];
    xs[t + 128] = x[((size_t)(n * CCH + t + 128)) * HWSZ + hw];
    __syncthreads();
    float acc = bfv[m * DD + t];
    const float* wp = WfT + (size_t)m * CCH * DD + t;
    #pragma unroll 4
    for (int c = 0; c < CCH; ++c) acc = fmaf(wp[(size_t)c * DD], xs[c], acc);
    unsigned short hi = f32_to_bf16(acc);
    size_t o = ((size_t)(m * NPOS + pos)) * DD + t;
    pre_hi[o] = hi;
    pre_lo[o] = f32_to_bf16(acc - bf16_to_f32(hi));
    red[t] = acc * acc;
    __syncthreads();
    for (int s = 64; s > 0; s >>= 1) { if (t < s) red[t] += red[t + s]; __syncthreads(); }
    if (t == 0) x2[m * NPOS + pos] = red[0];
}

// ---------------- main fused distance + gumbel + argmax kernel (WMMA) ----------------
// One wave handles TWO 16-position tiles (shared A loads -> 24 WMMAs per A tile) and
// one of SL K-slices. bf16 hi/lo split GEMM, f32 accumulate, fused gumbel + argmax.
__global__ __launch_bounds__(32) void k_dist_argmax(
    const unsigned short* __restrict__ pre_hi, const unsigned short* __restrict__ pre_lo,
    const unsigned short* __restrict__ mq_hi,  const unsigned short* __restrict__ mq_lo,
    const float* __restrict__ x2,  const float* __restrict__ c2,
    const float* __restrict__ expT, const float* __restrict__ gumbel,
    const float* __restrict__ temp,
    float* __restrict__ pval, int* __restrict__ pidx) {
    const int lane = threadIdx.x & 31;
    const int half = lane >> 4;
    const int l    = lane & 15;
    int bid   = blockIdx.x;          // [m(2)][ppair(128)][slice(SL)]
    int slice = bid & (SL - 1);
    int ppair = (bid >> 4) & 127;
    int m     = bid >> 11;
    int pos0  = ppair * 32 + l;      // tile 0 column owned by this lane
    int pos1  = pos0 + 16;           // tile 1 column

    // B operands: two position tiles, 4 d-chunks each, hi & lo (resident all K)
    V16 b0h[4], b0l[4], b1h[4], b1l[4];
    {
        const unsigned short* p0h = pre_hi + ((size_t)(m * NPOS + pos0)) * DD;
        const unsigned short* p0l = pre_lo + ((size_t)(m * NPOS + pos0)) * DD;
        const unsigned short* p1h = pre_hi + ((size_t)(m * NPOS + pos1)) * DD;
        const unsigned short* p1l = pre_lo + ((size_t)(m * NPOS + pos1)) * DD;
        #pragma unroll
        for (int cc = 0; cc < 4; ++cc) {
            int off = cc * 32 + half * 16;
            b0h[cc].q[0] = *(const uint4*)(p0h + off);
            b0h[cc].q[1] = *(const uint4*)(p0h + off + 8);
            b0l[cc].q[0] = *(const uint4*)(p0l + off);
            b0l[cc].q[1] = *(const uint4*)(p0l + off + 8);
            b1h[cc].q[0] = *(const uint4*)(p1h + off);
            b1h[cc].q[1] = *(const uint4*)(p1h + off + 8);
            b1l[cc].q[0] = *(const uint4*)(p1l + off);
            b1l[cc].q[1] = *(const uint4*)(p1l + off + 8);
        }
    }
    float x2v0 = x2[m * NPOS + pos0];
    float x2v1 = x2[m * NPOS + pos1];
    float invT = 1.0f / temp[0];

    const float* gbase0 = gumbel +
        ((size_t)(((pos0 >> 10) * MM + m) * HWSZ + (pos0 & (HWSZ - 1)))) * KK;
    const float* gbase1 = gumbel +
        ((size_t)(((pos1 >> 10) * MM + m) * HWSZ + (pos1 & (HWSZ - 1)))) * KK;

    float best0 = -3.4e38f, best1 = -3.4e38f;
    int   besti0 = 0, besti1 = 0;

    int kt0 = slice * KTPS;
    for (int kt = kt0; kt < kt0 + KTPS; ++kt) {
        int krow = kt * 16 + l;     // A-matrix row for this lane (both halves)
        const unsigned short* ah_p = mq_hi + ((size_t)(m * KK + krow)) * DD;
        const unsigned short* al_p = mq_lo + ((size_t)(m * KK + krow)) * DD;
        V16 ah[4], al[4];
        #pragma unroll
        for (int cc = 0; cc < 4; ++cc) {
            int offA = cc * 32 + half * 16;
            ah[cc].q[0] = *(const uint4*)(ah_p + offA);
            ah[cc].q[1] = *(const uint4*)(ah_p + offA + 8);
            al[cc].q[0] = *(const uint4*)(al_p + offA);
            al[cc].q[1] = *(const uint4*)(al_p + offA + 8);
        }
        v8f acc0 = {}, acc1 = {};
        #pragma unroll
        for (int cc = 0; cc < 4; ++cc) {
            acc0 = __builtin_amdgcn_wmma_f32_16x16x32_bf16(false, ah[cc].v, false, b0h[cc].v,
                                                           (short)0, acc0, false, false);
            acc1 = __builtin_amdgcn_wmma_f32_16x16x32_bf16(false, ah[cc].v, false, b1h[cc].v,
                                                           (short)0, acc1, false, false);
            acc0 = __builtin_amdgcn_wmma_f32_16x16x32_bf16(false, ah[cc].v, false, b0l[cc].v,
                                                           (short)0, acc0, false, false);
            acc1 = __builtin_amdgcn_wmma_f32_16x16x32_bf16(false, ah[cc].v, false, b1l[cc].v,
                                                           (short)0, acc1, false, false);
            acc0 = __builtin_amdgcn_wmma_f32_16x16x32_bf16(false, al[cc].v, false, b0h[cc].v,
                                                           (short)0, acc0, false, false);
            acc1 = __builtin_amdgcn_wmma_f32_16x16x32_bf16(false, al[cc].v, false, b1h[cc].v,
                                                           (short)0, acc1, false, false);
        }
        // epilogue: D VGPR r = k-row (kb + r), column = position l
        int kb = kt * 16 + half * 8;
        float4 c20 = *(const float4*)(c2 + m * KK + kb);
        float4 c21 = *(const float4*)(c2 + m * KK + kb + 4);
        float4 e0  = *(const float4*)(expT + m * KK + kb);
        float4 e1  = *(const float4*)(expT + m * KK + kb + 4);
        float4 ga0 = *(const float4*)(gbase0 + kb);
        float4 ga1 = *(const float4*)(gbase0 + kb + 4);
        float4 gb0 = *(const float4*)(gbase1 + kb);
        float4 gb1 = *(const float4*)(gbase1 + kb + 4);
        float cv[8] = {c20.x, c20.y, c20.z, c20.w, c21.x, c21.y, c21.z, c21.w};
        float ev[8] = {e0.x, e0.y, e0.z, e0.w, e1.x, e1.y, e1.z, e1.w};
        float g0[8] = {ga0.x, ga0.y, ga0.z, ga0.w, ga1.x, ga1.y, ga1.z, ga1.w};
        float g1[8] = {gb0.x, gb0.y, gb0.z, gb0.w, gb1.x, gb1.y, gb1.z, gb1.w};
        #pragma unroll
        for (int r = 0; r < 8; ++r) {
            int kidx = kb + r;
            float d0 = x2v0 + cv[r] - 2.0f * acc0[r];
            float s0 = (-d0 * ev[r] + g0[r]) * invT;
            if (s0 > best0 || (s0 == best0 && kidx < besti0)) { best0 = s0; besti0 = kidx; }
            float d1 = x2v1 + cv[r] - 2.0f * acc1[r];
            float s1 = (-d1 * ev[r] + g1[r]) * invT;
            if (s1 > best1 || (s1 == best1 && kidx < besti1)) { best1 = s1; besti1 = kidx; }
        }
    }
    // merge half-wave partner (same position column)
    float ov0 = __shfl_xor(best0, 16, 32);
    int   oi0 = __shfl_xor(besti0, 16, 32);
    if (ov0 > best0 || (ov0 == best0 && oi0 < besti0)) { best0 = ov0; besti0 = oi0; }
    float ov1 = __shfl_xor(best1, 16, 32);
    int   oi1 = __shfl_xor(besti1, 16, 32);
    if (ov1 > best1 || (ov1 == best1 && oi1 < besti1)) { best1 = ov1; besti1 = oi1; }
    if (half == 0) {
        size_t o0 = ((size_t)(m * NPOS + pos0)) * SL + slice;
        size_t o1 = ((size_t)(m * NPOS + pos1)) * SL + slice;
        pval[o0] = best0; pidx[o0] = besti0;
        pval[o1] = best1; pidx[o1] = besti1;
    }
}

// reduce the SL K-slice partials per (m,pos)
__global__ void k_argred(const float* __restrict__ pval, const int* __restrict__ pidx,
                         int* __restrict__ idxa) {
    int i = blockIdx.x * blockDim.x + threadIdx.x;
    if (i >= MM * NPOS) return;
    float bv = pval[(size_t)i * SL]; int bi = pidx[(size_t)i * SL];
    for (int s = 1; s < SL; ++s) {
        float v = pval[(size_t)i * SL + s]; int ix = pidx[(size_t)i * SL + s];
        if (v > bv || (v == bv && ix < bi)) { bv = v; bi = ix; }
    }
    idxa[i] = bi;
}

// out[n][o][hw] = table[0][idx0][o] + table[1][idx1][o] + btot[o]
__global__ __launch_bounds__(256) void k_out(const float* __restrict__ table,
                                             const float* __restrict__ btot,
                                             const int* __restrict__ idxa,
                                             float* __restrict__ out) {
    int bid = blockIdx.x;           // n*256 + o
    int n = bid >> 8, o = bid & 255;
    float bo = btot[o];
    for (int hw = threadIdx.x; hw < HWSZ; hw += 256) {
        int pos = n * HWSZ + hw;
        int i0 = idxa[pos];
        int i1 = idxa[NPOS + pos];
        float v = table[(size_t)i0 * CCH + o] +
                  table[((size_t)(KK + i1)) * CCH + o] + bo;
        out[((size_t)(n * CCH + o)) * HWSZ + hw] = v;
    }
}

extern "C" void kernel_launch(void* const* d_in, const int* in_sizes, int n_in,
                              void* d_out, int out_size, void* d_ws, size_t ws_size,
                              hipStream_t stream) {
    (void)in_sizes; (void)n_in; (void)out_size; (void)ws_size;
    const float* x        = (const float*)d_in[0];
    const float* temp     = (const float*)d_in[1];
    const float* gumbel   = (const float*)d_in[2];
    const float* codebook = (const float*)d_in[3];
    const float* wC_q     = (const float*)d_in[4];
    const float* wC_dq    = (const float*)d_in[5];
    const float* logT     = (const float*)d_in[6];
    const float* pre_w    = (const float*)d_in[7];
    const float* pre_b    = (const float*)d_in[8];
    const float* post_w   = (const float*)d_in[9];
    const float* post_b   = (const float*)d_in[10];
    const float* enc_w    = (const float*)d_in[11];
    const float* enc_b    = (const float*)d_in[12];
    const float* qh_w     = (const float*)d_in[13];
    const float* qh_b     = (const float*)d_in[14];
    const float* dqh_w    = (const float*)d_in[15];
    const float* dqh_b    = (const float*)d_in[16];
    const float* rest_w   = (const float*)d_in[17];
    const float* rest_b   = (const float*)d_in[18];
    float* out = (float*)d_out;

    char* wp = (char*)d_ws;
    auto alloc = [&](size_t bytes) {
        char* p = wp;
        wp += (bytes + 255) & ~(size_t)255;
        return p;
    };
    float*          T_     = (float*)alloc((size_t)CCH * CCH * 4);
    float*          qb2    = (float*)alloc(CCH * 4);
    float*          WfT    = (float*)alloc((size_t)MM * CCH * DD * 4);
    float*          bfv    = (float*)alloc(MM * DD * 4);
    unsigned short* mq_hi  = (unsigned short*)alloc((size_t)MM * KK * DD * 2);
    unsigned short* mq_lo  = (unsigned short*)alloc((size_t)MM * KK * DD * 2);
    float*          c2     = (float*)alloc(MM * KK * 4);
    float*          expT   = (float*)alloc(MM * KK * 4);
    float*          mdq    = (float*)alloc((size_t)MM * KK * DD * 4);
    float*          Wback  = (float*)alloc((size_t)CCH * CCH * 4);
    float*          G      = (float*)alloc((size_t)MM * CCH * DD * 4);
    float*          table  = (float*)alloc((size_t)MM * KK * CCH * 4);
    float*          btot   = (float*)alloc(CCH * 4);
    unsigned short* pre_hi = (unsigned short*)alloc((size_t)MM * NPOS * DD * 2);
    unsigned short* pre_lo = (unsigned short*)alloc((size_t)MM * NPOS * DD * 2);
    float*          x2     = (float*)alloc(MM * NPOS * 4);
    float*          pval   = (float*)alloc((size_t)MM * NPOS * SL * 4);
    int*            pidx   = (int*)alloc((size_t)MM * NPOS * SL * 4);
    int*            idxa   = (int*)alloc(MM * NPOS * 4);

    // ---- weight folding (small, f32 VALU) ----
    k_matmul256<<<CCH, CCH, 0, stream>>>(qh_w, enc_w, T_);          // T = qh_w @ enc_w
    k_qb2<<<1, CCH, 0, stream>>>(qh_w, enc_b, qh_b, qb2);
    k_wft<<<MM * CCH, DD, 0, stream>>>(pre_w, T_, WfT);
    k_bfv<<<1, MM * DD, 0, stream>>>(pre_w, qb2, pre_b, bfv);
    k_mapq<<<MM * KK, DD, 0, stream>>>(codebook, wC_q, mq_hi, mq_lo);
    k_c2<<<(MM * KK + 255) / 256, 256, 0, stream>>>(codebook, logT, c2, expT);
    k_mapdq<<<MM * KK, DD, 0, stream>>>(codebook, wC_dq, mdq);
    k_matmul256<<<CCH, CCH, 0, stream>>>(rest_w, dqh_w, Wback);     // Wback = rest_w @ dqh_w
    k_G<<<MM * CCH, DD, 0, stream>>>(Wback, post_w, G);
    k_table<<<MM * KK, CCH, 0, stream>>>(G, mdq, table);
    k_btot<<<1, CCH, 0, stream>>>(Wback, post_b, rest_w, dqh_b, rest_b, btot);

    // ---- per-pixel pipeline ----
    k_front<<<MM * NPOS, DD, 0, stream>>>(x, WfT, bfv, pre_hi, pre_lo, x2);
    k_dist_argmax<<<MM * 128 * SL, 32, 0, stream>>>(pre_hi, pre_lo, mq_hi, mq_lo,
                                                    x2, c2, expT, gumbel, temp,
                                                    pval, pidx);
    k_argred<<<(MM * NPOS + 255) / 256, 256, 0, stream>>>(pval, pidx, idxa);
    k_out<<<NN * CCH, 256, 0, stream>>>(table, btot, idxa, out);
}
